// CausalSelfAttention_2525440770211
// MI455X (gfx1250) — matrix operand order repaired and checked by reading, not compiled, not run on hardware
//
#include <hip/hip_runtime.h>
#include <math.h>

// Problem constants (match reference)
#define B_   2
#define T_   4096
#define C_   2048
#define H_   16
#define HK_  8
#define D_   128
#define GQA_ (H_ / HK_)
#define M_   (B_ * T_)      // 8192 rows

typedef __bf16 bf16;
typedef __attribute__((ext_vector_type(16))) __bf16 v16bf;
typedef __attribute__((ext_vector_type(8)))  __bf16 v8bf;
typedef __attribute__((ext_vector_type(4)))  __bf16 v4bf;
typedef __attribute__((ext_vector_type(8)))  float  v8f;

union V16U { v16bf v; v8bf h[2]; };

static __device__ inline v8f wmma_bf16(v16bf a, v16bf b, v8f c) {
  // 8 args: (neg_a, A, neg_b, B, c_mod, C, reuse_a, reuse_b)
  return __builtin_amdgcn_wmma_f32_16x16x32_bf16(false, a, false, b, (short)0, c,
                                                 false, false);
}

// ---------------------------------------------------------------------------
// fp32 -> bf16 conversion (4-wide)
// ---------------------------------------------------------------------------
__global__ __launch_bounds__(256)
void cvt_f32_bf16(const float4* __restrict__ src, v4bf* __restrict__ dst, int n4) {
  int stride = gridDim.x * blockDim.x;
  for (int i = blockIdx.x * blockDim.x + threadIdx.x; i < n4; i += stride) {
    float4 f = src[i];
    v4bf o;
    o.x = (bf16)f.x; o.y = (bf16)f.y; o.z = (bf16)f.z; o.w = (bf16)f.w;
    dst[i] = o;
  }
}

// ---------------------------------------------------------------------------
// GEMM: Cm[M,N] = A[M,K](bf16, row major) x W[N,K](bf16, row major)^T, fp32 out
// Block: 256 threads = 8 waves, 128(M) x 128(N) tile. Waves arranged 4(M)x2(N),
// each wave computes 32x64 via 2x4 WMMA accumulators. A and B k-tiles (32-wide)
// are staged in LDS with double buffering so global loads overlap WMMA.
// ---------------------------------------------------------------------------
__global__ __launch_bounds__(256)
void gemm_bf16_wmma(const bf16* __restrict__ A, const bf16* __restrict__ W,
                    float* __restrict__ Cm, int M, int N, int K) {
  __shared__ alignas(32) bf16 Abuf[2][128][32];  // 8 KB per buffer
  __shared__ alignas(32) bf16 Bbuf[2][128][32];  // 8 KB per buffer

  const int tid  = threadIdx.x;
  const int lane = tid & 31;
  const int wave = tid >> 5;
  const int col  = lane & 15;
  const int half = lane >> 4;
  const int wm   = wave & 3;   // M sub-block: 0..3 (x32 rows)
  const int wn   = wave >> 2;  // N sub-block: 0..1 (x64 cols)
  const int tileM = blockIdx.y * 128;
  const int tileN = blockIdx.x * 128;

  // cooperative global->LDS copy: thread t moves 32B of A and 32B of B per k-step
  const int crow  = tid >> 1;        // 0..127
  const int cpart = (tid & 1) * 16;  // 0 or 16 (k offset)

  const bf16* aSrc = A + (size_t)(tileM + crow) * K + cpart;
  const bf16* bSrc = W + (size_t)(tileN + crow) * K + cpart;

  // preload k-tile 0
  v16bf aReg = *(const v16bf*)aSrc;
  v16bf bReg = *(const v16bf*)bSrc;
  *(v16bf*)&Abuf[0][crow][cpart] = aReg;
  *(v16bf*)&Bbuf[0][crow][cpart] = bReg;
  __syncthreads();

  v8f acc[2][4] = {};

  for (int k0 = 0; k0 < K; k0 += 32) {
    const int buf   = (k0 >> 5) & 1;
    const bool more = (k0 + 32) < K;
    if (more) {  // issue next-tile global loads early; overlap with WMMA below
      aReg = *(const v16bf*)(aSrc + k0 + 32);
      bReg = *(const v16bf*)(bSrc + k0 + 32);
    }

    // A fragments (2 row subtiles) from LDS: lanes 0-15 K {0..15},
    // lanes 16-31 K {16..31}
    V16U af[2];
#pragma unroll
    for (int mt = 0; mt < 2; ++mt) {
      const int r = wm * 32 + mt * 16 + col;
      af[mt].h[0] = *(const v8bf*)&Abuf[buf][r][half * 16];
      af[mt].h[1] = *(const v8bf*)&Abuf[buf][r][half * 16 + 8];
    }
    // B fragments (4 col subtiles): lane col = N, 16 contiguous K per half
#pragma unroll
    for (int nt = 0; nt < 4; ++nt) {
      const int n = wn * 64 + nt * 16 + col;
      v16bf bfrag = *(const v16bf*)&Bbuf[buf][n][half * 16];
#pragma unroll
      for (int mt = 0; mt < 2; ++mt)
        acc[mt][nt] = wmma_bf16(af[mt].v, bfrag, acc[mt][nt]);
    }

    if (more) {
      *(v16bf*)&Abuf[buf ^ 1][crow][cpart] = aReg;
      *(v16bf*)&Bbuf[buf ^ 1][crow][cpart] = bReg;
    }
    __syncthreads();
  }

#pragma unroll
  for (int mt = 0; mt < 2; ++mt)
#pragma unroll
    for (int nt = 0; nt < 4; ++nt) {
      const int n = tileN + wn * 64 + nt * 16 + col;
#pragma unroll
      for (int r = 0; r < 8; ++r) {
        const int rr = tileM + wm * 32 + mt * 16 + r + half * 8;
        Cm[(size_t)rr * N + n] = acc[mt][nt][r];
      }
    }
}

// ---------------------------------------------------------------------------
// RoPE + RMS-norm, fp32 (bt, NH*D) -> bf16 (B, NH, T, D). One wave per (b,t,h).
// ---------------------------------------------------------------------------
__global__ __launch_bounds__(256)
void rope_rms_kernel(const float* __restrict__ src, const float* __restrict__ cosp,
                     const float* __restrict__ sinp, bf16* __restrict__ dst, int NH) {
  const int unit = blockIdx.x * 8 + (threadIdx.x >> 5);  // (b*T + t)*NH + h
  const int lane = threadIdx.x & 31;
  const int h  = unit % NH;
  const int bt = unit / NH;
  const int t  = bt % T_;
  const int b  = bt / T_;

  const float* s  = src  + (size_t)bt * NH * D_ + h * D_;
  const float* cp = cosp + (size_t)t * (D_ / 2);
  const float* sp = sinp + (size_t)t * (D_ / 2);

  float out[4];
#pragma unroll
  for (int j = 0; j < 2; ++j) {
    const int d = lane + 32 * j;           // 0..63
    const float x1 = s[d], x2 = s[d + 64];
    const float c = cp[d], sn = sp[d];
    out[j]     =  x1 * c  + x2 * sn;
    out[j + 2] = -x1 * sn + x2 * c;
  }
  float ss = out[0]*out[0] + out[1]*out[1] + out[2]*out[2] + out[3]*out[3];
#pragma unroll
  for (int m = 1; m < 32; m <<= 1) ss += __shfl_xor(ss, m, 32);
  const float inv = rsqrtf(ss * (1.0f / 128.0f) + 1.1920929e-7f);

  bf16* dp = dst + (((size_t)b * NH + h) * T_ + t) * D_;
  dp[lane]      = (bf16)(out[0] * inv);
  dp[lane + 32] = (bf16)(out[1] * inv);
  dp[lane + 64] = (bf16)(out[2] * inv);
  dp[lane + 96] = (bf16)(out[3] * inv);
}

// ---------------------------------------------------------------------------
// v = v + 2*sigmoid(x[:, :32] . wgate[hk]) * ve; write transposed (B,HK,D,T) bf16
// One wave per (b,t,hk).
// ---------------------------------------------------------------------------
__global__ __launch_bounds__(256)
void gate_v_kernel(const float* __restrict__ vf, const float* __restrict__ x,
                   const float* __restrict__ ve, const float* __restrict__ wgate,
                   bf16* __restrict__ vT) {
  const int unit = blockIdx.x * 8 + (threadIdx.x >> 5);  // (b*T + t)*HK + hk
  const int lane = threadIdx.x & 31;
  const int hk = unit % HK_;
  const int bt = unit / HK_;
  const int t  = bt % T_;
  const int b  = bt / T_;

  float g = x[(size_t)bt * C_ + lane] * wgate[hk * 32 + lane];
#pragma unroll
  for (int m = 1; m < 32; m <<= 1) g += __shfl_xor(g, m, 32);
  g = 2.0f / (1.0f + __expf(-g));

  const float* vp  = vf + (size_t)bt * (HK_ * D_) + hk * D_;
  const float* vep = ve + (size_t)bt * (HK_ * D_) + hk * D_;
  bf16* dp = vT + ((size_t)b * HK_ + hk) * (size_t)D_ * T_;
#pragma unroll
  for (int j = 0; j < 4; ++j) {
    const int d = lane + 32 * j;
    dp[(size_t)d * T_ + t] = (bf16)(vp[d] + g * vep[d]);
  }
}

// ---------------------------------------------------------------------------
// Sliding-window causal flash attention, bf16 WMMA, one wave per
// (b, h, 16-query block). Q:(B,H,T,D)  K:(B,HK,T,D)  V:(B,HK,D,T) all bf16.
// Y:(B*T, H*D) bf16.
// ---------------------------------------------------------------------------
__global__ __launch_bounds__(256)
void attn_wmma_kernel(const bf16* __restrict__ Q, const bf16* __restrict__ Kt,
                      const bf16* __restrict__ Vt, bf16* __restrict__ Y,
                      const int* __restrict__ wptr) {
  __shared__ alignas(32) bf16 pbuf[8][16 * 32];  // per-wave P staging (1 KB/wave)

  const int lane = threadIdx.x & 31;
  const int wave = threadIdx.x >> 5;
  const int col  = lane & 15;
  const int half = lane >> 4;

  const int unit = blockIdx.x * 8 + wave;       // (b*H + h)*(T/16) + qblk
  const int qblk = unit % (T_ / 16);
  const int bh   = unit / (T_ / 16);
  const int h    = bh % H_;
  const int b    = bh / H_;
  const int hk   = h / GQA_;
  const int W    = wptr[0];
  const int q0   = qblk * 16;

  const bf16* qbase = Q  + ((size_t)b * H_  + h)  * (size_t)T_ * D_;
  const bf16* kbase = Kt + ((size_t)b * HK_ + hk) * (size_t)T_ * D_;
  const bf16* vbase = Vt + ((size_t)b * HK_ + hk) * (size_t)D_ * T_;

  // Load Q fragments for all four 32-wide K-dim chunks of D=128 (held in regs)
  V16U qf[4];
  {
    const bf16* qr = qbase + (size_t)(q0 + col) * D_;
#pragma unroll
    for (int c = 0; c < 4; ++c) {
      const int dA = c * 32 + half * 16;
      qf[c].h[0] = *(const v8bf*)(qr + dA);
      qf[c].h[1] = *(const v8bf*)(qr + dA + 8);
    }
  }

  v8f o[8] = {};
  float mrow[8], lrow[8];
#pragma unroll
  for (int r = 0; r < 8; ++r) { mrow[r] = -INFINITY; lrow[r] = 0.f; }

  const float scale = 0.08838834764831845f;  // 1/sqrt(128)
  int kstart = q0 - W;
  if (kstart < 0) kstart = 0;
  kstart &= ~31;
  bf16* pw = pbuf[wave];

  for (int kk = kstart; kk <= q0 + 15; kk += 32) {
    // ---- S = Q K^T for 32 keys (two 16x16 tiles) ----
    v8f st[2];
#pragma unroll
    for (int t2 = 0; t2 < 2; ++t2) {
      v8f acc = {};
      const int key = kk + t2 * 16 + col;
      const bf16* kr = kbase + (size_t)key * D_;
#pragma unroll
      for (int c = 0; c < 4; ++c) {
        v16bf bfrag = *(const v16bf*)(kr + c * 32 + half * 16);
        acc = wmma_bf16(qf[c].v, bfrag, acc);
      }
#pragma unroll
      for (int r = 0; r < 8; ++r) {
        const int qrow = q0 + r + half * 8;
        const bool ok = (key <= qrow) && (key >= qrow - W);
        acc[r] = ok ? acc[r] * scale : -1e30f;
      }
      st[t2] = acc;
    }

    // ---- online softmax (row reductions across 16 lanes) ----
    float osc[8];
#pragma unroll
    for (int r = 0; r < 8; ++r) {
      float cm = fmaxf(st[0][r], st[1][r]);
#pragma unroll
      for (int m = 1; m < 16; m <<= 1) cm = fmaxf(cm, __shfl_xor(cm, m, 32));
      const float mn = fmaxf(mrow[r], cm);
      const float sc = __expf(mrow[r] - mn);
      const float p0 = __expf(st[0][r] - mn);
      const float p1 = __expf(st[1][r] - mn);
      float psum = p0 + p1;
#pragma unroll
      for (int m = 1; m < 16; m <<= 1) psum += __shfl_xor(psum, m, 32);
      lrow[r] = lrow[r] * sc + psum;
      mrow[r] = mn;
      osc[r]  = sc;
      // stage P in (q, key) row-major bf16 for A-fragment reload
      pw[(r + half * 8) * 32 + col]      = (bf16)p0;
      pw[(r + half * 8) * 32 + 16 + col] = (bf16)p1;
    }
#pragma unroll
    for (int dt = 0; dt < 8; ++dt)
#pragma unroll
      for (int r = 0; r < 8; ++r) o[dt][r] *= osc[r];

    // wave-private LDS: ensure stores land before A-fragment reload
    asm volatile("s_wait_dscnt 0" ::: "memory");

    V16U pf;
    pf.h[0] = *(const v8bf*)(pw + col * 32 + half * 16);
    pf.h[1] = *(const v8bf*)(pw + col * 32 + half * 16 + 8);

    // ---- O += P V over this 32-key chunk (V^T layout -> contiguous B frags) ----
#pragma unroll
    for (int dt = 0; dt < 8; ++dt) {
      const int dcol = dt * 16 + col;
      v16bf vfrag = *(const v16bf*)(vbase + (size_t)dcol * T_ + kk + half * 16);
      o[dt] = wmma_bf16(pf.v, vfrag, o[dt]);
    }
  }

  // ---- normalize and store y (B*T, H*D) bf16 ----
#pragma unroll
  for (int dt = 0; dt < 8; ++dt) {
#pragma unroll
    for (int r = 0; r < 8; ++r) {
      const int qrow = q0 + r + half * 8;
      Y[((size_t)b * T_ + qrow) * C_ + h * D_ + dt * 16 + col] =
          (bf16)(o[dt][r] / lrow[r]);
    }
  }
}

// ---------------------------------------------------------------------------
extern "C" void kernel_launch(void* const* d_in, const int* in_sizes, int n_in,
                              void* d_out, int out_size, void* d_ws, size_t ws_size,
                              hipStream_t stream) {
  const float* x     = (const float*)d_in[0];
  const float* ve    = (const float*)d_in[1];
  const float* cosp  = (const float*)d_in[2];
  const float* sinp  = (const float*)d_in[3];
  const float* wq    = (const float*)d_in[4];
  const float* wk    = (const float*)d_in[5];
  const float* wv    = (const float*)d_in[6];
  const float* wproj = (const float*)d_in[7];
  const float* wgate = (const float*)d_in[8];
  const int*   wsz   = (const int*)d_in[9];

  char* ws = (char*)d_ws;
  size_t off = 0;
  auto alloc = [&](size_t bytes) -> void* {
    void* p = ws + off;
    off = (off + bytes + 255) & ~(size_t)255;
    return p;
  };

  bf16*  xb  = (bf16*)alloc((size_t)M_ * C_ * 2);
  bf16*  wqb = (bf16*)alloc((size_t)(H_ * D_) * C_ * 2);
  bf16*  wkb = (bf16*)alloc((size_t)(HK_ * D_) * C_ * 2);
  bf16*  wvb = (bf16*)alloc((size_t)(HK_ * D_) * C_ * 2);
  bf16*  wpb = (bf16*)alloc((size_t)C_ * C_ * 2);
  float* qf  = (float*)alloc((size_t)M_ * H_ * D_ * 4);
  float* kf  = (float*)alloc((size_t)M_ * HK_ * D_ * 4);
  float* vf  = (float*)alloc((size_t)M_ * HK_ * D_ * 4);
  bf16*  qb  = (bf16*)alloc((size_t)M_ * H_ * D_ * 2);
  bf16*  kb  = (bf16*)alloc((size_t)M_ * HK_ * D_ * 2);
  bf16*  vT  = (bf16*)alloc((size_t)M_ * HK_ * D_ * 2);
  bf16*  yb  = (bf16*)alloc((size_t)M_ * C_ * 2);

  const dim3 blk(256);
  auto cvt_grid = [](size_t n4) {
    size_t g = (n4 + 255) / 256;
    return (unsigned)(g > 4096 ? 4096 : g);
  };

  // 1) fp32 -> bf16 conversions
  {
    size_t n;
    n = (size_t)M_ * C_ / 4;
    cvt_f32_bf16<<<cvt_grid(n), blk, 0, stream>>>((const float4*)x, (v4bf*)xb, (int)n);
    n = (size_t)(H_ * D_) * C_ / 4;
    cvt_f32_bf16<<<cvt_grid(n), blk, 0, stream>>>((const float4*)wq, (v4bf*)wqb, (int)n);
    n = (size_t)(HK_ * D_) * C_ / 4;
    cvt_f32_bf16<<<cvt_grid(n), blk, 0, stream>>>((const float4*)wk, (v4bf*)wkb, (int)n);
    cvt_f32_bf16<<<cvt_grid(n), blk, 0, stream>>>((const float4*)wv, (v4bf*)wvb, (int)n);
    n = (size_t)C_ * C_ / 4;
    cvt_f32_bf16<<<cvt_grid(n), blk, 0, stream>>>((const float4*)wproj, (v4bf*)wpb, (int)n);
  }

  // 2) QKV projection GEMMs (bf16 WMMA, fp32 out), 128x128 blocks
  gemm_bf16_wmma<<<dim3((H_ * D_) / 128, M_ / 128), blk, 0, stream>>>(
      xb, wqb, qf, M_, H_ * D_, C_);
  gemm_bf16_wmma<<<dim3((HK_ * D_) / 128, M_ / 128), blk, 0, stream>>>(
      xb, wkb, kf, M_, HK_ * D_, C_);
  gemm_bf16_wmma<<<dim3((HK_ * D_) / 128, M_ / 128), blk, 0, stream>>>(
      xb, wvb, vf, M_, HK_ * D_, C_);

  // 3) RoPE + RMS norm (q, k) -> bf16 tiled layouts
  rope_rms_kernel<<<(M_ * H_) / 8, blk, 0, stream>>>(qf, cosp, sinp, qb, H_);
  rope_rms_kernel<<<(M_ * HK_) / 8, blk, 0, stream>>>(kf, cosp, sinp, kb, HK_);

  // 4) gate + ve add, write V transposed
  gate_v_kernel<<<(M_ * HK_) / 8, blk, 0, stream>>>(vf, x, ve, wgate, vT);

  // 5) sliding-window flash attention
  attn_wmma_kernel<<<(B_ * H_ * (T_ / 16)) / 8, blk, 0, stream>>>(qb, kb, vT, yb, wsz);

  // 6) output projection -> fp32 d_out
  gemm_bf16_wmma<<<dim3(C_ / 128, M_ / 128), blk, 0, stream>>>(
      yb, wpb, (float*)d_out, M_, C_, C_);
}
